// BlockModel_33002528703065
// MI455X (gfx1250) — matrix operand order repaired and checked by reading, not compiled, not run on hardware
//
#include <hip/hip_runtime.h>

// ---------------- problem dims ----------------
#define TT 128   // sequence length
#define CC 64    // channels
#define HH 8     // heads
#define HSZ 8    // head size

typedef _Float16 half16 __attribute__((ext_vector_type(16)));
typedef float    float8 __attribute__((ext_vector_type(8)));

// ---------------- LDS layout (element strides chosen for bank-conflict avoidance) ----------------
#define XS   68    // xbuf row stride (floats): rows 272B -> 16B aligned for async b128 DMA
#define HBS  66    // hbuf   row stride (halves)  : dword bank = (33*t + j) % 64, 33 odd -> distinct
#define ABS_ 66    // attnbuf row stride (halves)
#define QS   68    // qbuf/kbuf row stride (halves): dword bank = 2*(17*t mod 32) -> distinct for 16 lanes
#define VTS  136   // v-transposed [64][136] halves
#define PS   132   // per-wave P strip [16][132] halves
#define US   260   // ffn hidden [128][260] halves

#define XB_OFF  0
#define HB_OFF  (XB_OFF + TT*XS*4)            // 34816
#define AB_OFF  (HB_OFF + TT*HBS*2)           // 51712
#define BIG_OFF (AB_OFF + TT*ABS_*2)          // 68608
#define QB_OFF  BIG_OFF
#define KB_OFF  (QB_OFF + TT*QS*2)            // +17408
#define VT_OFF  (KB_OFF + TT*QS*2)            // +17408
#define PB_OFF  (VT_OFF + CC*VTS*2)           // +17408
#define UB_OFF  BIG_OFF                        // FFN hidden overlays Q/K/Vt/P (phase-disjoint)
#define SMEM_BYTES (PB_OFF + 4*16*PS*2)        // 137728 bytes -> 2 workgroups / 320KB WGP

// ---------------- WMMA helper ----------------
static __device__ inline float8 wmma_f16(half16 a, half16 b, float8 c) {
    // D = A(16x32 f16) * B(32x16 f16) + C(16x16 f32)
    return __builtin_amdgcn_wmma_f32_16x16x32_f16(false, a, false, b, (short)0, c, false, false);
}

// K index held by (vgpr v, lane half lhi) in a 16-bit A fragment (contiguous order, matches B)
static __device__ inline int a_kk(int v, int lhi) {
    return 16 * lhi + 2 * v;
}

// A fragment (16x32) from LDS, row-major [row][k], all 32 k valid. strides/offsets must be even.
static __device__ inline half16 load_a(const _Float16* base, int stride, int r0, int c0,
                                       int lrow, int lhi) {
    union { half16 h; unsigned u[8]; } f;
    const unsigned* p = (const unsigned*)base;
    int rowoff = (r0 + lrow) * stride + c0;
#pragma unroll
    for (int v = 0; v < 8; ++v)
        f.u[v] = p[(rowoff + a_kk(v, lhi)) >> 1];
    return f.h;
}

// A fragment with only k<8 valid (Q over head-size dim), rest zero-padded.
static __device__ inline half16 load_a_k8(const _Float16* base, int stride, int r0, int c0,
                                          int lrow, int lhi) {
    union { half16 h; unsigned u[8]; } f;
    const unsigned* p = (const unsigned*)base;
    int rowoff = (r0 + lrow) * stride + c0;
#pragma unroll
    for (int v = 0; v < 8; ++v) {
        int kk = a_kk(v, lhi);
        f.u[v] = (kk < 8) ? p[(rowoff + kk) >> 1] : 0u;
    }
    return f.h;
}

// B fragment (32x16) from transposed layout [n][k]; lane holds column n = n0+lrow,
// k = 16*lhi + 2v (+1). Contiguous 32B per lane -> two b128 loads (global weights).
static __device__ inline half16 load_bt_global(const _Float16* base, int stride, int n0, int k0,
                                               int lrow, int lhi) {
    union { half16 h; uint4 q[2]; } f;
    const uint4* p = (const uint4*)(base + (n0 + lrow) * stride + k0 + 16 * lhi);
    f.q[0] = p[0];
    f.q[1] = p[1];
    return f.h;
}

// B fragment from transposed LDS layout [n][k] with per-lane validity (used for V^T, n<8 valid).
static __device__ inline half16 load_bt_lds(const _Float16* base, int stride, int n0, int k0,
                                            int lrow, int lhi, bool valid) {
    union { half16 h; unsigned u[8]; } f;
    const unsigned* p = (const unsigned*)base;
    int rowoff = (n0 + lrow) * stride + k0 + 16 * lhi;
#pragma unroll
    for (int v = 0; v < 8; ++v)
        f.u[v] = valid ? p[(rowoff + 2 * v) >> 1] : 0u;
    return f.h;
}

// B fragment from row-major K buffer [t][c], contraction over head dim (k<8 valid).
static __device__ inline half16 load_b_k8(const _Float16* base, int stride, int n0, int c0,
                                          int lrow, int lhi) {
    union { half16 h; unsigned u[8]; } f;
    const unsigned* p = (const unsigned*)base;
    int rowoff = (n0 + lrow) * stride + c0;
#pragma unroll
    for (int v = 0; v < 8; ++v) {
        int kk = 16 * lhi + 2 * v;
        f.u[v] = (kk < 8) ? p[(rowoff + kk) >> 1] : 0u;
    }
    return f.h;
}

// ---------------- weight prep: fp32 -> f16, transposed to [n][k]; fold 1/sqrt(HS) into Wq ----------------
__global__ void prep_weights(const float* __restrict__ Wq, const float* __restrict__ Wk,
                             const float* __restrict__ Wv, const float* __restrict__ Wp,
                             const float* __restrict__ W1, const float* __restrict__ W2,
                             _Float16* __restrict__ ws) {
    int i = blockIdx.x * 256 + threadIdx.x;   // 49152 total
    if (i < 12288) {                          // WqT / WkT / WvT : [n=h*8+d][c]
        int g = i >> 12, r = i & 4095;
        int n = r >> 6, c = r & 63;
        int h = n >> 3, d = n & 7;
        const float* W = (g == 0) ? Wq : (g == 1) ? Wk : Wv;
        float v = W[(h * 64 + c) * 8 + d];
        if (g == 0) v *= 0.35355339059327373f;           // 1/sqrt(8)
        ws[i] = (_Float16)v;
    } else if (i < 16384) {                   // WpT [n][c]
        int r = i - 12288, n = r >> 6, c = r & 63;
        ws[i] = (_Float16)Wp[c * 64 + n];
    } else if (i < 32768) {                   // W1T [n=256][c=64]
        int r = i - 16384, n = r >> 6, c = r & 63;
        ws[i] = (_Float16)W1[c * 256 + n];
    } else if (i < 49152) {                   // W2T [n=64][k=256]
        int r = i - 32768, n = r >> 8, k = r & 255;
        ws[i] = (_Float16)W2[k * 64 + n];
    }
}

// ---------------- fused transformer block: one workgroup per batch element ----------------
__global__ void __launch_bounds__(128)
block_fused(const float* __restrict__ x, const _Float16* __restrict__ wbuf,
            const float* __restrict__ bp, const float* __restrict__ b1,
            const float* __restrict__ b2, const float* __restrict__ g1,
            const float* __restrict__ be1, const float* __restrict__ g2,
            const float* __restrict__ be2, float* __restrict__ out) {
    extern __shared__ unsigned char smem[];
    float*    xbuf   = (float*)(smem + XB_OFF);
    _Float16* hbuf   = (_Float16*)(smem + HB_OFF);
    _Float16* attnb  = (_Float16*)(smem + AB_OFF);
    _Float16* qbuf   = (_Float16*)(smem + QB_OFF);
    _Float16* kbuf   = (_Float16*)(smem + KB_OFF);
    _Float16* vtbuf  = (_Float16*)(smem + VT_OFF);
    _Float16* pbuf   = (_Float16*)(smem + PB_OFF);
    _Float16* ubuf   = (_Float16*)(smem + UB_OFF);

    const _Float16* wqkvT = wbuf;            // 3 x [64][64]
    const _Float16* wpT   = wbuf + 12288;    // [64][64]
    const _Float16* w1T   = wbuf + 16384;    // [256][64]
    const _Float16* w2T   = wbuf + 32768;    // [64][256]

    const int tid  = threadIdx.x;
    const int wid  = tid >> 5;
    const int lane = tid & 31;
    const int lrow = lane & 15;
    const int lhi  = lane >> 4;
    const int b    = blockIdx.x;

    const float* xg = x + (size_t)b * (TT * CC);

    // ---- phase 1: async DMA x tile [128][64] global -> LDS (ASYNCcnt-tracked, no VGPR round-trip) ----
    {
        const unsigned xbase = (unsigned)(size_t)(void*)xbuf;   // LDS byte address of xbuf
#pragma unroll
        for (int k = 0; k < 16; ++k) {
            int chunk = tid + k * 128;                 // 2048 x 16B chunks, coalesced
            int row = chunk >> 4, c4 = chunk & 15;
            unsigned ldsoff = xbase + row * (XS * 4) + c4 * 16;      // 16B-aligned (XS=68)
            unsigned long long ga =
                (unsigned long long)(const void*)(xg + row * CC + c4 * 4);
            asm volatile("global_load_async_to_lds_b128 %0, %1, off"
                         :: "v"(ldsoff), "v"(ga) : "memory");
        }
        // warm L2 with the 96KB f16 weight pack while the async DMA is in flight
#pragma unroll
        for (int j = 0; j < 3; ++j)
            __builtin_prefetch((const char*)wbuf + (tid + j * 128) * 256, 0, 3);
        asm volatile("s_wait_asynccnt 0" ::: "memory");
    }
    __syncthreads();

    // ---- phase 2: LayerNorm 1 (one row per thread) ----
    {
        const float* row = xbuf + tid * XS;
        float s = 0.f, s2 = 0.f;
        for (int c = 0; c < CC; ++c) { float v = row[c]; s += v; s2 += v * v; }
        float mu = s * (1.0f / CC);
        float rstd = rsqrtf(s2 * (1.0f / CC) - mu * mu + 1e-5f);
        for (int c = 0; c < CC; ++c)
            hbuf[tid * HBS + c] = (_Float16)((row[c] - mu) * rstd * g1[c] + be1[c]);
    }
    __syncthreads();

    // ---- phase 3: QKV = h @ W{q,k,v}'  (3 x 128x64x64) ----
    for (int tile = wid; tile < 96; tile += 4) {
        int g = tile / 32, rem = tile % 32, rt = rem >> 2, ct = rem & 3;
        const _Float16* WT = wqkvT + g * 4096;
        float8 acc = {};
#pragma unroll
        for (int kc = 0; kc < 2; ++kc) {
            half16 a  = load_a(hbuf, HBS, rt * 16, kc * 32, lrow, lhi);
            half16 bf = load_bt_global(WT, 64, ct * 16, kc * 32, lrow, lhi);
            acc = wmma_f16(a, bf, acc);
        }
        union { float8 v; float f[8]; } o; o.v = acc;
        if (g == 2) {                                   // store V transposed [feature][t]
#pragma unroll
            for (int v = 0; v < 8; ++v)
                vtbuf[(ct * 16 + lrow) * VTS + rt * 16 + v + 8 * lhi] = (_Float16)o.f[v];
        } else {
            _Float16* dst = (g == 0) ? qbuf : kbuf;
#pragma unroll
            for (int v = 0; v < 8; ++v)
                dst[(rt * 16 + v + 8 * lhi) * QS + ct * 16 + lrow] = (_Float16)o.f[v];
        }
    }
    __syncthreads();

    // ---- phase 4: attention, (head, row-strip) pairs round-robin over waves ----
    for (int pair = wid; pair < 64; pair += 4) {
        int h = pair & 7, strip = pair >> 3, r0 = strip * 16;
        half16 qf = load_a_k8(qbuf, QS, r0, h * 8, lrow, lhi);   // scale folded into Wq
        float s[8][8];
#pragma unroll
        for (int j = 0; j < 8; ++j) {
            half16 kf = load_b_k8(kbuf, QS, j * 16, h * 8, lrow, lhi);
            float8 acc = {};
            acc = wmma_f16(qf, kf, acc);
            union { float8 v; float f[8]; } o; o.v = acc;
#pragma unroll
            for (int v = 0; v < 8; ++v) {
                int row = r0 + v + 8 * lhi, col = j * 16 + lrow;
                s[j][v] = (col <= row) ? o.f[v] : -3.0e38f;      // causal mask
            }
        }
        // row softmax: in-lane over j, cross-lane over 16-lane row groups
        float inv[8];
#pragma unroll
        for (int v = 0; v < 8; ++v) {
            float m = s[0][v];
#pragma unroll
            for (int j = 1; j < 8; ++j) m = fmaxf(m, s[j][v]);
            for (int off = 1; off < 16; off <<= 1) m = fmaxf(m, __shfl_xor(m, off, 32));
            float su = 0.f;
#pragma unroll
            for (int j = 0; j < 8; ++j) { float p = __expf(s[j][v] - m); s[j][v] = p; su += p; }
            for (int off = 1; off < 16; off <<= 1) su += __shfl_xor(su, off, 32);
            inv[v] = 1.0f / su;
        }
        // stage P (f16) in wave-private LDS strip, then wait on DScnt before reuse as A
        _Float16* prow = pbuf + wid * 16 * PS;
#pragma unroll
        for (int j = 0; j < 8; ++j)
#pragma unroll
            for (int v = 0; v < 8; ++v)
                prow[(v + 8 * lhi) * PS + j * 16 + lrow] = (_Float16)(s[j][v] * inv[v]);
        asm volatile("s_wait_dscnt 0" ::: "memory");

        float8 oacc = {};
#pragma unroll
        for (int kc = 0; kc < 4; ++kc) {
            half16 pf = load_a(prow, PS, 0, kc * 32, lrow, lhi);
            half16 vf = load_bt_lds(vtbuf, VTS, h * 8, kc * 32, lrow, lhi, lrow < 8);
            oacc = wmma_f16(pf, vf, oacc);
        }
        union { float8 v; float f[8]; } o; o.v = oacc;
        if (lrow < 8) {
#pragma unroll
            for (int v = 0; v < 8; ++v)
                attnb[(r0 + v + 8 * lhi) * ABS_ + h * 8 + lrow] = (_Float16)o.f[v];
        }
    }
    __syncthreads();

    // ---- phase 5: x2 = x + attn @ Wp + bp (in-place into xbuf) ----
    for (int tile = wid; tile < 32; tile += 4) {
        int rt = tile >> 2, ct = tile & 3;
        float8 acc = {};
#pragma unroll
        for (int kc = 0; kc < 2; ++kc) {
            half16 a  = load_a(attnb, ABS_, rt * 16, kc * 32, lrow, lhi);
            half16 bf = load_bt_global(wpT, 64, ct * 16, kc * 32, lrow, lhi);
            acc = wmma_f16(a, bf, acc);
        }
        union { float8 v; float f[8]; } o; o.v = acc;
#pragma unroll
        for (int v = 0; v < 8; ++v) {
            int m = rt * 16 + v + 8 * lhi, n = ct * 16 + lrow;
            xbuf[m * XS + n] += o.f[v] + bp[n];
        }
    }
    __syncthreads();

    // ---- phase 6: LayerNorm 2 ----
    {
        const float* row = xbuf + tid * XS;
        float s = 0.f, s2 = 0.f;
        for (int c = 0; c < CC; ++c) { float v = row[c]; s += v; s2 += v * v; }
        float mu = s * (1.0f / CC);
        float rstd = rsqrtf(s2 * (1.0f / CC) - mu * mu + 1e-5f);
        for (int c = 0; c < CC; ++c)
            hbuf[tid * HBS + c] = (_Float16)((row[c] - mu) * rstd * g2[c] + be2[c]);
    }
    __syncthreads();

    // ---- phase 7: U = relu(h2 @ W1 + b1)  [128x256] ----
    for (int tile = wid; tile < 128; tile += 4) {
        int rt = tile >> 4, ct = tile & 15;
        float8 acc = {};
#pragma unroll
        for (int kc = 0; kc < 2; ++kc) {
            half16 a  = load_a(hbuf, HBS, rt * 16, kc * 32, lrow, lhi);
            half16 bf = load_bt_global(w1T, 64, ct * 16, kc * 32, lrow, lhi);
            acc = wmma_f16(a, bf, acc);
        }
        union { float8 v; float f[8]; } o; o.v = acc;
#pragma unroll
        for (int v = 0; v < 8; ++v) {
            int m = rt * 16 + v + 8 * lhi, n = ct * 16 + lrow;
            ubuf[m * US + n] = (_Float16)fmaxf(o.f[v] + b1[n], 0.0f);
        }
    }
    __syncthreads();

    // ---- phase 8: out = x2 + U @ W2 + b2 ----
    float* og = out + (size_t)b * (TT * CC);
    for (int tile = wid; tile < 32; tile += 4) {
        int rt = tile >> 2, ct = tile & 3;
        float8 acc = {};
#pragma unroll
        for (int kc = 0; kc < 8; ++kc) {
            half16 a  = load_a(ubuf, US, kc * 0 + rt * 16, kc * 32, lrow, lhi);
            half16 bf = load_bt_global(w2T, 256, ct * 16, kc * 32, lrow, lhi);
            acc = wmma_f16(a, bf, acc);
        }
        union { float8 v; float f[8]; } o; o.v = acc;
#pragma unroll
        for (int v = 0; v < 8; ++v) {
            int m = rt * 16 + v + 8 * lhi, n = ct * 16 + lrow;
            og[m * CC + n] = xbuf[m * XS + n] + o.f[v] + b2[n];
        }
    }
}

extern "C" void kernel_launch(void* const* d_in, const int* in_sizes, int n_in,
                              void* d_out, int out_size, void* d_ws, size_t ws_size,
                              hipStream_t stream) {
    (void)in_sizes; (void)n_in; (void)out_size; (void)ws_size;
    const float* x   = (const float*)d_in[0];
    const float* Wq  = (const float*)d_in[1];
    const float* Wk  = (const float*)d_in[2];
    const float* Wv  = (const float*)d_in[3];
    const float* Wp  = (const float*)d_in[4];
    const float* bp  = (const float*)d_in[5];
    const float* W1  = (const float*)d_in[6];
    const float* b1  = (const float*)d_in[7];
    const float* W2  = (const float*)d_in[8];
    const float* b2  = (const float*)d_in[9];
    const float* g1  = (const float*)d_in[10];
    const float* be1 = (const float*)d_in[11];
    const float* g2  = (const float*)d_in[12];
    const float* be2 = (const float*)d_in[13];
    float* outp = (float*)d_out;
    _Float16* wbuf = (_Float16*)d_ws;

    // allow >64KB dynamic LDS (gfx1250: 320KB/WGP); deterministic, capture-safe host call
    (void)hipFuncSetAttribute((const void*)block_fused,
                              hipFuncAttributeMaxDynamicSharedMemorySize, SMEM_BYTES);

    prep_weights<<<192, 256, 0, stream>>>(Wq, Wk, Wv, Wp, W1, W2, wbuf);
    block_fused<<<2048, 128, SMEM_BYTES, stream>>>(x, wbuf, bp, b1, b2, g1, be1, g2, be2, outp);
}
